// PairStateLayer_69123203662020
// MI455X (gfx1250) — matrix operand order repaired and checked by reading, not compiled, not run on hardware
//
#include <hip/hip_runtime.h>
#include <hip/hip_bf16.h>

// ---------------------------------------------------------------------------
// PairStateLayer on MI455X (gfx1250, wave32).
// GEMMs: V_WMMA_F32_16X16X32_BF16 with double-buffered LDS tiles staged by
// GLOBAL_LOAD_ASYNC_TO_LDS_B128 (ASYNCcnt) -- no VGPR round trip for tiles.
// The sequential scan keeps the 64x64 pair state in LDS (VALU kernel),
// split over 4 e-column chunks per (b,h) for 256-way block parallelism.
// ---------------------------------------------------------------------------

typedef __attribute__((ext_vector_type(16))) __bf16        v16bf;
typedef __attribute__((ext_vector_type(8)))  float         v8f;
typedef __attribute__((ext_vector_type(8)))  unsigned short ushort8;

union Frag {
    ushort8 h[2];
    v16bf   v;
};

__device__ __forceinline__ unsigned short f32_to_bf16(float f) {
    union { float f; unsigned int u; } v; v.f = f;
    unsigned int r = v.u + 0x7FFFu + ((v.u >> 16) & 1u);   // round-to-nearest-even
    return (unsigned short)(r >> 16);
}
__device__ __forceinline__ float bf16_to_f32(unsigned short u) {
    union { unsigned int u; float f; } v; v.u = ((unsigned int)u) << 16;
    return v.f;
}

// Async copy of one 64-byte row (32 bf16) from global to LDS.
// ISA: each op copies 16B from global memory into LDS
// (offset applies to BOTH addresses), tracked by ASYNCcnt.
__device__ __forceinline__ void async_copy_row64(unsigned lds, const unsigned short* g) {
    asm volatile(
        "global_load_async_to_lds_b128 %0, %1, off\n\t"
        "global_load_async_to_lds_b128 %0, %1, off offset:16\n\t"
        "global_load_async_to_lds_b128 %0, %1, off offset:32\n\t"
        "global_load_async_to_lds_b128 %0, %1, off offset:48"
        :: "v"(lds), "v"(g) : "memory");
}
__device__ __forceinline__ void wait_async0() {
    asm volatile("s_wait_asynccnt 0x0" ::: "memory");
}

// ---------------------------------------------------------------------------
// f32 -> bf16 conversion (weights)
// ---------------------------------------------------------------------------
__global__ __launch_bounds__(256)
void cvt_bf16_kernel(const float* __restrict__ src, unsigned short* __restrict__ dst, int n) {
    int i = blockIdx.x * 256 + threadIdx.x;
    if (i < n) dst[i] = f32_to_bf16(src[i]);
}

// ---------------------------------------------------------------------------
// LayerNorm over DM=512, output bf16. One block (256 thr) per row.
// ---------------------------------------------------------------------------
__global__ __launch_bounds__(256)
void layernorm_bf16_kernel(const float* __restrict__ x, const float* __restrict__ g,
                           const float* __restrict__ be, unsigned short* __restrict__ out) {
    const int DM = 512;
    int row = blockIdx.x;
    int t = threadIdx.x;
    const float* xr = x + (size_t)row * DM;
    float v0 = xr[t], v1 = xr[t + 256];
    float s = v0 + v1;
    float q = v0 * v0 + v1 * v1;
#pragma unroll
    for (int off = 16; off > 0; off >>= 1) {      // wave32 reduce
        s += __shfl_xor(s, off);
        q += __shfl_xor(q, off);
    }
    __shared__ float rs[8], rq[8];
    int w = t >> 5, lane = t & 31;
    if (lane == 0) { rs[w] = s; rq[w] = q; }
    __syncthreads();
    float S = 0.f, Q = 0.f;
#pragma unroll
    for (int i = 0; i < 8; ++i) { S += rs[i]; Q += rq[i]; }
    float mean = S * (1.f / 512.f);
    float var  = Q * (1.f / 512.f) - mean * mean;
    float inv  = rsqrtf(var + 1e-5f);
    unsigned short* orow = out + (size_t)row * DM;
    orow[t]       = f32_to_bf16((v0 - mean) * inv * g[t] + be[t]);
    orow[t + 256] = f32_to_bf16((v1 - mean) * inv * g[t + 256] + be[t + 256]);
}

// ---------------------------------------------------------------------------
// WMMA GEMM, NT form:  C[m,n] = sum_k A[m,k] * W[n,k]
//   A: [M,K] bf16 row-major, W: [N,K] bf16 row-major.
// Block tile 128(M) x 64(N), 8 waves, wave tile 32x32 (2x2 WMMA tiles).
// Tiles double-buffered in LDS, filled by async global->LDS b128 copies.
// Epilogues: 0=tanh->f32   1=+resid->f32   2=+bias,gelu->bf16   3=+bias,+resid->f32
// ---------------------------------------------------------------------------
#define EPI_TANH       0
#define EPI_RESID      1
#define EPI_BIAS_GELU  2
#define EPI_BIAS_RESID 3

template <int EPI>
__global__ __launch_bounds__(256)
void wmma_gemm_kernel(const unsigned short* __restrict__ A,
                      const unsigned short* __restrict__ W,
                      float* __restrict__ Cf, unsigned short* __restrict__ Cb,
                      const float* __restrict__ bias, const float* __restrict__ resid,
                      int M, int N, int K) {
    // Double-buffered LDS tiles, padded rows (stride 40 ushort = 80B, rows 16B-aligned).
    __shared__ alignas(16) unsigned short As[2][128][40];
    __shared__ alignas(16) unsigned short Ws[2][64][40];

    const int t    = threadIdx.x;
    const int wv   = t >> 5;          // wave 0..7
    const int lane = t & 31;
    const int wm   = wv >> 1;         // 0..3  (M direction)
    const int wn   = wv & 1;          // 0..1  (N direction)
    const int m0   = blockIdx.y * 128;
    const int n0   = blockIdx.x * 64;
    const int r16  = lane & 15;
    const int hi   = lane >> 4;       // 0 or 1

    // Staging assignment: threads 0..127 -> A rows, 128..191 -> W rows.
    // One thread stages a full 32-bf16 (64B) row per K-step, 4 async b128 ops.
    bool stager = false;
    const unsigned short* gsrc = nullptr;
    unsigned ldsrow[2] = {0u, 0u};
    if (t < 128) {
        stager = true;
        gsrc = A + (size_t)(m0 + t) * K;
        ldsrow[0] = (unsigned)(uintptr_t)&As[0][t][0];
        ldsrow[1] = (unsigned)(uintptr_t)&As[1][t][0];
    } else if (t < 192) {
        stager = true;
        int rw = t - 128;
        gsrc = W + (size_t)(n0 + rw) * K;
        ldsrow[0] = (unsigned)(uintptr_t)&Ws[0][rw][0];
        ldsrow[1] = (unsigned)(uintptr_t)&Ws[1][rw][0];
    }

    v8f acc[2][2] = {};

    // Prologue: stage tile 0 into buffer 0.
    if (stager) async_copy_row64(ldsrow[0], gsrc);
    wait_async0();
    __syncthreads();

    int cur = 0;
    for (int kb = 0; kb < K; kb += 32) {
        // Prefetch next K-tile into the back buffer (overlaps with compute below).
        if (kb + 32 < K && stager) async_copy_row64(ldsrow[cur ^ 1], gsrc + kb + 32);

        // ---- fragments per ISA 7.12.2 layouts ----
        // A (16x32, 16-bit): lane<16 holds row M=lane, K {0..7,16..23}; lane>=16: K {8..15,24..31}
        // B (32x16, 16-bit): lane<16 holds col N=lane, K {0..7,16..23}; lane>=16: K {8..15,24..31}
        const unsigned short (*Ab)[40] = As[cur];
        const unsigned short (*Bb)[40] = Ws[cur];
        Frag fa[2], fb[2];
        const int ka  = hi << 3;      // 0 or 8
        const int kbb = hi << 4;      // 0 or 16
#pragma unroll
        for (int i = 0; i < 2; ++i) {
            int arow = wm * 32 + i * 16 + r16;
            fa[i].h[0] = *(const ushort8*)&Ab[arow][ka];
            fa[i].h[1] = *(const ushort8*)&Ab[arow][ka + 16];
            int brow = wn * 32 + i * 16 + r16;
            fb[i].h[0] = *(const ushort8*)&Bb[brow][ka];
            fb[i].h[1] = *(const ushort8*)&Bb[brow][ka + 16];
        }
#pragma unroll
        for (int i = 0; i < 2; ++i)
#pragma unroll
            for (int j = 0; j < 2; ++j)
                acc[i][j] = __builtin_amdgcn_wmma_f32_16x16x32_bf16(
                    false, fa[i].v, false, fb[j].v, (short)0, acc[i][j], false, false);

        // Next tile must be fully landed in LDS; buffer we just read is free to
        // be overwritten only after everyone passes the barrier.
        wait_async0();
        __syncthreads();
        cur ^= 1;
    }

    // ---- epilogue + store; C/D layout: VGPR i -> (M = i or 8+i, N = lane&15) ----
#pragma unroll
    for (int i = 0; i < 2; ++i) {
#pragma unroll
        for (int j = 0; j < 2; ++j) {
#pragma unroll
            for (int e = 0; e < 8; ++e) {
                int mrow = m0 + wm * 32 + i * 16 + ((lane < 16) ? e : e + 8);
                int ncol = n0 + wn * 32 + j * 16 + (lane & 15);
                size_t idx = (size_t)mrow * N + ncol;
                float v = acc[i][j][e];
                if constexpr (EPI == EPI_TANH) {
                    Cf[idx] = tanhf(v);
                } else if constexpr (EPI == EPI_RESID) {
                    Cf[idx] = v + resid[idx];
                } else if constexpr (EPI == EPI_BIAS_GELU) {
                    v += bias[ncol];
                    v = 0.5f * v * (1.0f + erff(v * 0.70710678118654752f));
                    Cb[idx] = f32_to_bf16(v);
                } else {
                    Cf[idx] = v + bias[ncol] + resid[idx];
                }
            }
        }
    }
}

// ---------------------------------------------------------------------------
// sd/pd gates: sigmoid(n @ Wsd^T + bsd), sigmoid(n @ Wpd^T + bpd). N=8 each.
// One block (256 thr) per row: 16 outputs x 16-lane dot groups over K=512.
// ---------------------------------------------------------------------------
__global__ __launch_bounds__(256)
void sdpd_kernel(const unsigned short* __restrict__ nb,
                 const float* __restrict__ Wsd, const float* __restrict__ bsd,
                 const float* __restrict__ Wpd, const float* __restrict__ bpd,
                 float* __restrict__ sd, float* __restrict__ pd) {
    const int K = 512;
    int row = blockIdx.x;
    int t = threadIdx.x;
    int o = t >> 4;                    // 0..15 (0..7 -> sd, 8..15 -> pd)
    int g = t & 15;
    const unsigned short* nr = nb + (size_t)row * K;
    const float* wr = (o < 8) ? (Wsd + (size_t)o * K) : (Wpd + (size_t)(o - 8) * K);
    float acc = 0.f;
    for (int k = g; k < K; k += 16)
        acc += bf16_to_f32(nr[k]) * wr[k];
#pragma unroll
    for (int off = 8; off > 0; off >>= 1) acc += __shfl_down(acc, off, 16);
    if (g == 0) {
        float bi = (o < 8) ? bsd[o] : bpd[o - 8];
        float v  = 1.f / (1.f + expf(-(acc + bi)));
        if (o < 8) sd[(size_t)row * 8 + o]      = v;
        else       pd[(size_t)row * 8 + o - 8]  = v;
    }
}

// ---------------------------------------------------------------------------
// Sequential scan. Block = one (b, h, e-chunk of 16 columns). pair in LDS.
// proj layout: [B,S,2048] f32, columns: [0,512)=a  [512,1024)=b  [1024,1536)=ql
// [1536,2048)=qr, each organized as [H=8][SD=64].
// Emits lcq = (lc * qr) as bf16 [B,S,512] (A operand of the Wout GEMM).
// ---------------------------------------------------------------------------
__global__ __launch_bounds__(256)
void scan_kernel(const float* __restrict__ proj, const float* __restrict__ sd_all,
                 const float* __restrict__ pd_all, unsigned short* __restrict__ lcq) {
    const int S = 2048, CH = 16;
    int blk   = blockIdx.x;            // 0..255
    int bh    = blk >> 2;
    int chunk = blk & 3;
    int b = bh >> 3, h = bh & 7;
    int e_off = chunk * CH;

    __shared__ float pair[64][CH + 1]; // [d][e_local], padded
    __shared__ float state[64];
    __shared__ float avec[64], qlvec[64], bvec[CH];
    __shared__ float red[16][16];      // [d-group][e_local]

    int t = threadIdx.x;
    for (int i = t; i < 64 * (CH + 1); i += 256) (&pair[0][0])[i] = 0.f;
    if (t < 64) state[t] = 0.f;
    __syncthreads();

    const int d4   = t >> 2;           // 0..63  (pair-update row)
    const int esub = (t & 3) << 2;     // 4 e-columns each
    const int eL   = t & 15;           // lc mapping
    const int g    = t >> 4;           // 16 d-groups of 4

    for (int s = 0; s < S; ++s) {
        const float* base = proj + ((size_t)b * S + s) * 2048 + h * 64;
        if (t < 64) { avec[t] = base[t]; qlvec[t] = base[1024 + t]; }
        else if (t < 64 + CH) bvec[t - 64] = base[512 + e_off + (t - 64)];
        float sdv = sd_all[((size_t)b * S + s) * 8 + h];
        float pdv = pd_all[((size_t)b * S + s) * 8 + h];
        __syncthreads();

        // pair = pd*pair + (1-pd) * outer(prev_state, b)
        float wgt = (1.f - pdv) * state[d4];
#pragma unroll
        for (int j = 0; j < 4; ++j) {
            int e = esub + j;
            pair[d4][e] = pdv * pair[d4][e] + wgt * bvec[e];
        }
        __syncthreads();

        // state = sd*state + (1-sd)*a   (after pair used prev state)
        if (t < 64) state[t] = sdv * state[t] + (1.f - sdv) * avec[t];

        // lc[e] = sum_d pair[d][e] * ql[d]
        float acc = 0.f;
#pragma unroll
        for (int j = 0; j < 4; ++j) {
            int d = (g << 2) + j;
            acc += pair[d][eL] * qlvec[d];
        }
        red[g][eL] = acc;
        __syncthreads();
        if (t < CH) {
            float lc = 0.f;
#pragma unroll
            for (int gg = 0; gg < 16; ++gg) lc += red[gg][t];
            float qr = base[1536 + e_off + t];
            lcq[((size_t)b * S + s) * 512 + h * 64 + e_off + t] = f32_to_bf16(lc * qr);
        }
        __syncthreads();
    }
}

// ---------------------------------------------------------------------------
// Host-side orchestration (graph-capture safe: launches only, on `stream`).
// ---------------------------------------------------------------------------
extern "C" void kernel_launch(void* const* d_in, const int* in_sizes, int n_in,
                              void* d_out, int out_size, void* d_ws, size_t ws_size,
                              hipStream_t stream) {
    (void)in_sizes; (void)n_in; (void)out_size; (void)ws_size;

    const float* x    = (const float*)d_in[0];
    const float* ng   = (const float*)d_in[1];
    const float* nbet = (const float*)d_in[2];
    const float* fg   = (const float*)d_in[3];
    const float* fbet = (const float*)d_in[4];
    const float* Wa   = (const float*)d_in[5];
    const float* Wb   = (const float*)d_in[6];
    const float* Wql  = (const float*)d_in[7];
    const float* Wqr  = (const float*)d_in[8];
    const float* Wsd  = (const float*)d_in[9];
    const float* bsd  = (const float*)d_in[10];
    const float* Wpd  = (const float*)d_in[11];
    const float* bpd  = (const float*)d_in[12];
    const float* Wout = (const float*)d_in[13];
    const float* W1   = (const float*)d_in[14];
    const float* b1   = (const float*)d_in[15];
    const float* W2   = (const float*)d_in[16];
    const float* b2   = (const float*)d_in[17];
    float* out = (float*)d_out;

    const size_t M = 16384;   // B*S

    // Workspace layout (~200 MB, 256B-aligned chunks).
    char* ws = (char*)d_ws;
    size_t off = 0;
    auto alloc = [&](size_t bytes) -> void* {
        void* p = ws + off;
        off = (off + bytes + 255) & ~(size_t)255;
        return p;
    };
    unsigned short* nbf   = (unsigned short*)alloc(M * 512 * 2);      // LN(x); reused as LN(r)
    unsigned short* Wcat  = (unsigned short*)alloc(2048 * 512 * 2);   // [Wa;Wb;Wql;Wqr]
    unsigned short* Woutb = (unsigned short*)alloc(512 * 512 * 2);
    unsigned short* W1b   = (unsigned short*)alloc(2048 * 512 * 2);
    unsigned short* W2b   = (unsigned short*)alloc(512 * 2048 * 2);
    float*          proj  = (float*)alloc(M * 2048 * 4);              // reused as h1 (bf16)
    float*          sdb   = (float*)alloc(M * 8 * 4);
    float*          pdb   = (float*)alloc(M * 8 * 4);
    unsigned short* lcq   = (unsigned short*)alloc(M * 512 * 2);
    float*          r     = (float*)alloc(M * 512 * 4);
    unsigned short* h1    = (unsigned short*)proj;                    // alias (proj consumed by scan)

    // 1) n = LN(x) -> bf16
    layernorm_bf16_kernel<<<M, 256, 0, stream>>>(x, ng, nbet, nbf);

    // 2) weight conversions to bf16
    const int QW = 512 * 512;
    cvt_bf16_kernel<<<(QW + 255) / 256, 256, 0, stream>>>(Wa,  Wcat,          QW);
    cvt_bf16_kernel<<<(QW + 255) / 256, 256, 0, stream>>>(Wb,  Wcat + QW,     QW);
    cvt_bf16_kernel<<<(QW + 255) / 256, 256, 0, stream>>>(Wql, Wcat + 2 * QW, QW);
    cvt_bf16_kernel<<<(QW + 255) / 256, 256, 0, stream>>>(Wqr, Wcat + 3 * QW, QW);
    cvt_bf16_kernel<<<(QW + 255) / 256, 256, 0, stream>>>(Wout, Woutb, QW);
    const int W1N = 2048 * 512;
    cvt_bf16_kernel<<<(W1N + 255) / 256, 256, 0, stream>>>(W1, W1b, W1N);
    cvt_bf16_kernel<<<(W1N + 255) / 256, 256, 0, stream>>>(W2, W2b, W1N);

    // 3) proj = tanh(n @ Wcat^T)   [16384 x 2048]
    wmma_gemm_kernel<EPI_TANH><<<dim3(2048 / 64, M / 128), 256, 0, stream>>>(
        nbf, Wcat, proj, nullptr, nullptr, nullptr, (int)M, 2048, 512);

    // 4) gates
    sdpd_kernel<<<M, 256, 0, stream>>>(nbf, Wsd, bsd, Wpd, bpd, sdb, pdb);

    // 5) sequential scan -> lcq (bf16)
    scan_kernel<<<256, 256, 0, stream>>>(proj, sdb, pdb, lcq);

    // 6) r = x + lcq @ Wout^T
    wmma_gemm_kernel<EPI_RESID><<<dim3(512 / 64, M / 128), 256, 0, stream>>>(
        lcq, Woutb, r, nullptr, nullptr, x, (int)M, 512, 512);

    // 7) n2 = LN(r) -> bf16 (reuse nbf)
    layernorm_bf16_kernel<<<M, 256, 0, stream>>>(r, fg, fbet, nbf);

    // 8) h1 = gelu(n2 @ W1^T + b1) -> bf16  (reuses proj buffer)
    wmma_gemm_kernel<EPI_BIAS_GELU><<<dim3(2048 / 64, M / 128), 256, 0, stream>>>(
        nbf, W1b, nullptr, h1, b1, nullptr, (int)M, 2048, 512);

    // 9) out = r + h1 @ W2^T + b2
    wmma_gemm_kernel<EPI_BIAS_RESID><<<dim3(512 / 64, M / 128), 256, 0, stream>>>(
        h1, W2b, out, nullptr, b2, r, (int)M, 512, 2048);
}
